// DFlashSelfAttention_64381559767753
// MI455X (gfx1250) — compile-verified, matrix operand order checked by reading, not run on hardware
//
#include <hip/hip_runtime.h>
#include <hip/hip_bf16.h>

// ---------------------------------------------------------------------------
// Problem constants (from reference)
// ---------------------------------------------------------------------------
#define B_SZ 2
#define L_SZ 2048
#define HID 4096
#define NH 16
#define NKV 4
#define HD 64
#define QKV_COLS ((NH + 2 * NKV) * HD) // 1536
#define ATT_COLS (NH * HD)             // 1024
#define ROWS (B_SZ * L_SZ)             // 4096

typedef __attribute__((ext_vector_type(16))) __bf16 v16bf;
typedef __attribute__((ext_vector_type(8))) __bf16 v8bf;
typedef __attribute__((ext_vector_type(8))) float v8f;

// CDNA5 16-bit A/B fragment K-index for element e of a lane:
//   lanes 0-15 (half=0):  e=0..7 -> K=e,      e=8..15 -> K=16+(e-8)
//   lanes 16-31 (half=1): e=0..7 -> K=8+e,    e=8..15 -> K=24+(e-8)
// Inverse (pack side): hi=(k>>3)&1, e=(k&7)|(((k>>4)&1)<<3), lane=(idx&15)+16*hi
__device__ __forceinline__ int kmap(int e, int half) {
  return (e & 7) + ((e >> 3) << 4) + (half << 3);
}

// Native wave32 XOR lane-swap via ds_swizzle (group-of-32 mode:
// offset = {1'b0, xor[14:10], or[9:5], and[4:0]}).
template <int IMM>
__device__ __forceinline__ float swzf(float x) {
  return __int_as_float(__builtin_amdgcn_ds_swizzle(__float_as_int(x), IMM));
}
#define SWZ_XOR1 0x041f
#define SWZ_XOR2 0x081f
#define SWZ_XOR4 0x101f
#define SWZ_XOR8 0x201f
#define SWZ_XOR16 0x401f

// ---------------------------------------------------------------------------
// Pack fp32 A[M][K] (row-major) -> bf16 fragment-major
// P[(mtile*(K/32)+kc)*32 + lane][16 elems].  One thread = 8 k-elems of a row
// (32B coalesced read, one 16B contiguous write).
// ---------------------------------------------------------------------------
template <int M, int K>
__global__ __launch_bounds__(256) void pack_a_kernel(
    const float* __restrict__ A, __bf16* __restrict__ P) {
  constexpr int KG = K / 8;
  const int gid = blockIdx.x * 256 + threadIdx.x; // < M*KG
  const int row = gid / KG;
  const int k0 = (gid % KG) * 8;
  const float4 f0 = *(const float4*)(A + (size_t)row * K + k0);
  const float4 f1 = *(const float4*)(A + (size_t)row * K + k0 + 4);
  const int kc = k0 >> 5;
  const int kk0 = k0 & 31;
  const int hi = (kk0 >> 3) & 1;
  const int eb = ((kk0 >> 4) & 1) << 3; // element base 0 or 8
  const int lp = (row & 15) + 16 * hi;
  __bf16 o[8] = {(__bf16)f0.x, (__bf16)f0.y, (__bf16)f0.z, (__bf16)f0.w,
                 (__bf16)f1.x, (__bf16)f1.y, (__bf16)f1.z, (__bf16)f1.w};
  __bf16* dst =
      P + (((size_t)((row >> 4) * (K / 32) + kc) * 32 + lp) * 16 + eb);
  *(v8bf*)dst = *(v8bf*)o;
}

// ---------------------------------------------------------------------------
// Pack fp32 B[K][N] (row-major) -> bf16 fragment-major
// P[(ntile*(K/32)+kc)*32 + lane][16 elems].  One thread = rows (k,k+1) x 8
// consecutive cols (coalesced reads, 8 merged-b32 scattered writes).
// ---------------------------------------------------------------------------
template <int K, int N>
__global__ __launch_bounds__(256) void pack_b_kernel(
    const float* __restrict__ B, __bf16* __restrict__ P) {
  constexpr int CG = N / 8;
  const int gid = blockIdx.x * 256 + threadIdx.x; // < (K/2)*CG
  const int k = (gid / CG) * 2;
  const int c0 = (gid % CG) * 8;
  const float* b0 = B + (size_t)k * N + c0;
  const float4 g0 = *(const float4*)(b0);
  const float4 g1 = *(const float4*)(b0 + 4);
  const float4 g2 = *(const float4*)(b0 + N);
  const float4 g3 = *(const float4*)(b0 + N + 4);
  const float r0[8] = {g0.x, g0.y, g0.z, g0.w, g1.x, g1.y, g1.z, g1.w};
  const float r1[8] = {g2.x, g2.y, g2.z, g2.w, g3.x, g3.y, g3.z, g3.w};
  const int kc = k >> 5;
  const int kk = k & 31;
  const int hi = (kk >> 3) & 1;
  const int e = (kk & 7) + (((kk >> 4) & 1) << 3); // even
#pragma unroll
  for (int j = 0; j < 8; ++j) {
    const int col = c0 + j;
    const int lp = (col & 15) + 16 * hi;
    __bf16* dst =
        P + (((size_t)((col >> 4) * (K / 32) + kc) * 32 + lp) * 16 + e);
    dst[0] = (__bf16)r0[j]; // k   -> element e   (merged b32 store)
    dst[1] = (__bf16)r1[j]; // k+1 -> element e+1
  }
}

// ---------------------------------------------------------------------------
// LDS-free, barrier-free GEMM on pre-packed bf16 fragments (f32 accumulate).
// C[M,N] = A[M,K] @ B[K,N].  Operands are L2-resident (192MB L2 >> A,B);
// each lane's fragment is 32 contiguous bytes, lane stride 32B -> fully
// coalesced b128 pairs straight into WMMA operands.  Workgroup tile 128x128,
// 8 independent waves of 32x64; double-buffered K loop (KC even).
// ---------------------------------------------------------------------------
template <int M, int N, int K>
__global__ __launch_bounds__(256) void gemm_packed_wmma(
    const __bf16* __restrict__ Ap, const __bf16* __restrict__ Bp,
    float* __restrict__ C) {
  constexpr int KC = K / 32;
  static_assert((KC & 1) == 0, "KC must be even");
  const int tid = threadIdx.x;
  const int lane = tid & 31;
  const int wv = tid >> 5;
  const int hi = lane >> 4;
  const int c16 = lane & 15;
  const int m0 = blockIdx.y * 128;
  const int n0 = blockIdx.x * 128;
  const int amt = (m0 >> 4) + (wv & 3) * 2;  // A mtile base (2 tiles)
  const int bnt = (n0 >> 4) + (wv >> 2) * 4; // B ntile base (4 tiles)

  const __bf16* __restrict__ pA0 = Ap + (size_t)(amt + 0) * KC * 512 + lane * 16;
  const __bf16* __restrict__ pA1 = Ap + (size_t)(amt + 1) * KC * 512 + lane * 16;
  const __bf16* __restrict__ pB0 = Bp + (size_t)(bnt + 0) * KC * 512 + lane * 16;
  const __bf16* __restrict__ pB1 = Bp + (size_t)(bnt + 1) * KC * 512 + lane * 16;
  const __bf16* __restrict__ pB2 = Bp + (size_t)(bnt + 2) * KC * 512 + lane * 16;
  const __bf16* __restrict__ pB3 = Bp + (size_t)(bnt + 3) * KC * 512 + lane * 16;

  v8f acc[2][4] = {};
  v16bf af[2], bf[4], ag[2], bg[4];

  auto load_frags = [&](v16bf(&a)[2], v16bf(&b)[4], int kc) {
    const size_t o = (size_t)kc * 512;
    a[0] = *(const v16bf*)(pA0 + o);
    a[1] = *(const v16bf*)(pA1 + o);
    b[0] = *(const v16bf*)(pB0 + o);
    b[1] = *(const v16bf*)(pB1 + o);
    b[2] = *(const v16bf*)(pB2 + o);
    b[3] = *(const v16bf*)(pB3 + o);
  };
  auto do_wmma = [&](const v16bf(&a)[2], const v16bf(&b)[4]) {
#pragma unroll
    for (int i = 0; i < 2; ++i)
#pragma unroll
      for (int j = 0; j < 4; ++j)
        acc[i][j] = __builtin_amdgcn_wmma_f32_16x16x32_bf16(
            false, a[i], false, b[j], (short)0, acc[i][j], false, false);
  };

  load_frags(af, bf, 0);
  for (int kc = 0; kc < KC; kc += 2) {
    load_frags(ag, bg, kc + 1);
    if (kc + 4 < KC) { // warm L2 ahead
      __builtin_prefetch(pA0 + (size_t)(kc + 4) * 512, 0, 0);
      __builtin_prefetch(pB0 + (size_t)(kc + 4) * 512, 0, 0);
    }
    do_wmma(af, bf);
    if (kc + 2 < KC) load_frags(af, bf, kc + 2);
    do_wmma(ag, bg);
  }

  // Epilogue: C/D layout -> VGPR v: lanes 0-15 M=v, lanes 16-31 M=v+8.
  // Compile-time N => single base address + immediate-offset stores.
  float* __restrict__ cp =
      C + (size_t)(m0 + (wv & 3) * 32 + 8 * hi) * N + n0 + (wv >> 2) * 64 + c16;
#pragma unroll
  for (int i = 0; i < 2; ++i)
#pragma unroll
    for (int j = 0; j < 4; ++j)
#pragma unroll
      for (int v = 0; v < 8; ++v)
        cp[(size_t)(i * 16 + v) * N + j * 16] = acc[i][j][v];
}

// ---------------------------------------------------------------------------
// Block-local attention: 4-wave workgroup per (batch, kv-group, 16-token
// block); the 4 heads of a GQA group share K/V staged once in LDS.
// RMSNorm + RoPE in fp32, QK^T and P@V via bf16 WMMA, causal-in-block mask.
// Output is written DIRECTLY in packed-fragment bf16 layout for GEMM2's A.
// ---------------------------------------------------------------------------
__global__ __launch_bounds__(128) void attn16_kernel(
    const float* __restrict__ qkv,  // [B*L, 1536] fp32
    const float* __restrict__ cosT, // [B, L, 64]
    const float* __restrict__ sinT, // [B, L, 64]
    const float* __restrict__ qnw,  // [64]
    const float* __restrict__ knw,  // [64]
    __bf16* __restrict__ apack) {   // packed A for GEMM2 (M=B*L, K=1024)
  __shared__ alignas(16) float qlds[4][16][64];
  __shared__ alignas(16) float klds[16][64];
  __shared__ alignas(16) float vlds[16][64];
  __shared__ alignas(16) float plds[4][16][16];

  const int tid = threadIdx.x;
  const int w = tid >> 5;  // wave -> head within kv group
  const int lane = tid & 31;
  const int r = lane & 15; // row within 16-block (also frag column id)
  const int hi = lane >> 4;
  const int qb = blockIdx.x;  // 0..127 token block
  const int kvg = blockIdx.y; // 0..3 kv group
  const int b = blockIdx.z;   // 0..1 batch
  const int h = kvg * 4 + w;  // this wave's query head

  const int t = qb * 16 + r;
  const size_t rowbase = ((size_t)b * L_SZ + t) * QKV_COLS;
  const float* cp = cosT + ((size_t)b * L_SZ + t) * HD + hi * 32;
  const float* sp = sinT + ((size_t)b * L_SZ + t) * HD + hi * 32;
  const float sgn = hi ? 1.f : -1.f; // rotate_half sign

  // --- Q for this wave's head: RMSNorm + RoPE ---
  {
    const float* qp = qkv + rowbase + h * HD + hi * 32;
    float x[32];
    float ss = 0.f;
#pragma unroll
    for (int i = 0; i < 32; ++i) {
      x[i] = qp[i];
      ss += x[i] * x[i];
    }
    ss += swzf<SWZ_XOR16>(ss); // combine row halves (lanes r, r+16)
    const float rs = rsqrtf(ss * (1.f / (float)HD) + 1e-6f);
#pragma unroll
    for (int i = 0; i < 32; ++i) {
      const int d = hi * 32 + i;
      const float xn = x[i] * rs * qnw[d];
      const float part = swzf<SWZ_XOR16>(xn); // x[d +/- 32]
      qlds[w][r][d] = xn * cp[i] + sgn * part * sp[i];
    }
  }
  // --- wave 0: K (RMSNorm + RoPE); wave 1: V copy ---
  if (w == 0) {
    const float* kp = qkv + rowbase + (NH + kvg) * HD + hi * 32;
    float x[32];
    float ss = 0.f;
#pragma unroll
    for (int i = 0; i < 32; ++i) {
      x[i] = kp[i];
      ss += x[i] * x[i];
    }
    ss += swzf<SWZ_XOR16>(ss);
    const float rs = rsqrtf(ss * (1.f / (float)HD) + 1e-6f);
#pragma unroll
    for (int i = 0; i < 32; ++i) {
      const int d = hi * 32 + i;
      const float xn = x[i] * rs * knw[d];
      const float part = swzf<SWZ_XOR16>(xn);
      klds[r][d] = xn * cp[i] + sgn * part * sp[i];
    }
  } else if (w == 1) {
    const float* vp = qkv + rowbase + (NH + NKV + kvg) * HD + hi * 32;
#pragma unroll
    for (int i = 0; i < 32; ++i) vlds[r][hi * 32 + i] = vp[i];
  }
  __syncthreads();

  // --- scores = Q @ K^T via 2 bf16 WMMAs (K-dim 64) ---
  v16bf qa0, qa1, kb0, kb1;
#pragma unroll
  for (int e = 0; e < 16; ++e) {
    const int kk = kmap(e, hi);
    qa0[e] = (__bf16)qlds[w][r][kk];
    qa1[e] = (__bf16)qlds[w][r][32 + kk];
    kb0[e] = (__bf16)klds[r][kk]; // lane holds key-row r => K^T in B slot
    kb1[e] = (__bf16)klds[r][32 + kk];
  }
  v8f sc = {};
  sc = __builtin_amdgcn_wmma_f32_16x16x32_bf16(false, qa0, false, kb0,
                                               (short)0, sc, false, false);
  sc = __builtin_amdgcn_wmma_f32_16x16x32_bf16(false, qa1, false, kb1,
                                               (short)0, sc, false, false);

  // --- per-row softmax with causal-in-block mask ---
#pragma unroll
  for (int ve = 0; ve < 8; ++ve) {
    const int m = ve + 8 * hi;
    const int n = r;
    float sv = (n <= m) ? sc[ve] * 0.125f : -3.0e38f; // 1/sqrt(64), mask
    float mx = sv;
    mx = fmaxf(mx, swzf<SWZ_XOR1>(mx));
    mx = fmaxf(mx, swzf<SWZ_XOR2>(mx));
    mx = fmaxf(mx, swzf<SWZ_XOR4>(mx));
    mx = fmaxf(mx, swzf<SWZ_XOR8>(mx));
    const float ex = __expf(sv - mx);
    float sm = ex;
    sm += swzf<SWZ_XOR1>(sm);
    sm += swzf<SWZ_XOR2>(sm);
    sm += swzf<SWZ_XOR4>(sm);
    sm += swzf<SWZ_XOR8>(sm);
    plds[w][m][n] = ex / sm;
  }
  // per-wave private plds slice: no cross-wave barrier needed here.

  // --- out = P @ V (P zero-padded to 16x32, V to 32x64): 4 WMMAs ---
  v16bf pa;
  v16bf vb[4];
#pragma unroll
  for (int e = 0; e < 16; ++e) {
    const int kk = kmap(e, hi);
    pa[e] = (kk < 16) ? (__bf16)plds[w][r][kk] : (__bf16)0.f;
#pragma unroll
    for (int nt = 0; nt < 4; ++nt)
      vb[nt][e] = (kk < 16) ? (__bf16)vlds[kk][nt * 16 + r] : (__bf16)0.f;
  }

  // Output row block = 16-aligned: mtile = b*128 + qb.  Column (gemm2 K dim)
  // col = h*64 + nt*16 + r -> kc = h*2 + (nt>>1), e = (r&7)+8*(nt&1),
  // packed lane = m + 16*((r>>3)&1).
  const int mtile2 = b * (L_SZ / 16) + qb;
#pragma unroll
  for (int nt = 0; nt < 4; ++nt) {
    v8f o = {};
    o = __builtin_amdgcn_wmma_f32_16x16x32_bf16(false, pa, false, vb[nt],
                                                (short)0, o, false, false);
    const int kc2 = h * 2 + (nt >> 1);
    const int ep = (r & 7) + 8 * (nt & 1);
    const int hip = (r >> 3) & 1;
#pragma unroll
    for (int ve = 0; ve < 8; ++ve) {
      const int mrow = ve + 8 * hi;
      apack[(((size_t)mtile2 * (ATT_COLS / 32) + kc2) * 32 + (mrow + 16 * hip)) *
                16 +
            ep] = (__bf16)o[ve];
    }
  }
}

// ---------------------------------------------------------------------------
// Launch: pack(hidden), pack(Wqkv) -> GEMM1 -> attention (emits packed A2)
//         -> pack(Wo) -> GEMM2.  All stream-ordered.
// Workspace layout (bytes):
//   [0)            qkv fp32       ROWS*1536*4 = 25,165,824
//   [25165824)     Apack region   ROWS*HID*2  = 33,554,432 (reused: attn out)
//   [58720256)     Bpack region   HID*1536*2  = 12,582,912 (reused: Wo pack)
// ---------------------------------------------------------------------------
extern "C" void kernel_launch(void* const* d_in, const int* in_sizes, int n_in,
                              void* d_out, int out_size, void* d_ws,
                              size_t ws_size, hipStream_t stream) {
  const float* hidden = (const float*)d_in[0];
  // d_in[1] = attention_mask (block-causal) — applied analytically, not read.
  const float* cosT = (const float*)d_in[2];
  const float* sinT = (const float*)d_in[3];
  const float* Wqkv = (const float*)d_in[4];
  const float* Wo = (const float*)d_in[5];
  const float* qnw = (const float*)d_in[6];
  const float* knw = (const float*)d_in[7];
  float* out = (float*)d_out;

  char* ws = (char*)d_ws;
  float* qkv_ws = (float*)ws;                            // ROWS x 1536 fp32
  __bf16* apack = (__bf16*)(ws + (size_t)25165824);      // packed A region
  __bf16* bpack = (__bf16*)(ws + (size_t)58720256);      // packed B region

  // Pack inputs for GEMM1 (bandwidth-trivial, ~138MB total traffic).
  pack_a_kernel<ROWS, HID>
      <<<(ROWS * (HID / 8)) / 256, 256, 0, stream>>>(hidden, apack);
  pack_b_kernel<HID, QKV_COLS>
      <<<((HID / 2) * (QKV_COLS / 8)) / 256, 256, 0, stream>>>(Wqkv, bpack);

  dim3 g1(QKV_COLS / 128, ROWS / 128);
  gemm_packed_wmma<ROWS, QKV_COLS, HID>
      <<<g1, 256, 0, stream>>>(apack, bpack, qkv_ws);

  // Attention writes GEMM2's A operand directly in packed bf16 (reuses the
  // Apack region — dead after GEMM1).
  dim3 gAt(L_SZ / 16, NKV, B_SZ);
  attn16_kernel<<<gAt, 128, 0, stream>>>(qkv_ws, cosT, sinT, qnw, knw, apack);

  // Wo pack reuses the Bpack region (dead after GEMM1).
  pack_b_kernel<ATT_COLS, HID>
      <<<((ATT_COLS / 2) * (HID / 8)) / 256, 256, 0, stream>>>(Wo, bpack);

  dim3 g2(HID / 128, ROWS / 128);
  gemm_packed_wmma<ROWS, HID, ATT_COLS>
      <<<g2, 256, 0, stream>>>(apack, bpack, out);
}
